// TransformerBlock_87411174408370
// MI455X (gfx1250) — matrix operand order repaired and checked by reading, not compiled, not run on hardware
//
#include <hip/hip_runtime.h>

// ---------------- types / constants ----------------
typedef __bf16 bf16;
typedef __attribute__((ext_vector_type(16))) __bf16 v16bf;
typedef __attribute__((ext_vector_type(8)))  float  v8f;
typedef unsigned int u32x4 __attribute__((ext_vector_type(4)));
typedef int          i32x8 __attribute__((ext_vector_type(8)));
typedef int          i32x4 __attribute__((ext_vector_type(4)));

#define HDIM   768
#define HEADS  12
#define HD     64
#define INTER  3072
#define BATCH  4
#define TSEQ   1024
#define MTOK   4096          // BATCH*TSEQ
#define EPSV   1e-6f

__device__ __forceinline__ bf16  f2bf(float f) { return (bf16)f; }
__device__ __forceinline__ float bf2f(bf16 b)  { return (float)b; }

__device__ __forceinline__ unsigned lds_off_of(const void* p) {
    // low 32 bits of a generic pointer into LDS == LDS byte offset (aperture rule)
    return (unsigned)(unsigned long long)(uintptr_t)p;
}

// ---------------- Tensor Data Mover: 2D strided tile -> LDS ----------------
// D# flags (group1 word0): data_size=2B (bit16) | pad_enable (bit20)
//                          | pad_interval<<22 | pad_amount<<25
#define TDM_FLAGS(padi, pada) ((1u << 16) | (1u << 20) | ((padi) << 22) | ((pada) << 25))
#define TDM_FLAGS_NOPAD        (1u << 16)

__device__ __forceinline__ void tdm_load_2d(
    unsigned lds_off, const void* gptr, unsigned flags,
    unsigned tile_d0, unsigned tile_d1,
    unsigned tensor_d0, unsigned tensor_d1,
    unsigned long long stride0)   // elements
{
    unsigned long long ga = (unsigned long long)(uintptr_t)gptr;
    u32x4 g0;
    g0[0] = 1u;                                   // count=1, user descriptor
    g0[1] = lds_off;                              // LDS byte address
    g0[2] = (unsigned)ga;                         // global addr [31:0]
    g0[3] = (unsigned)((ga >> 32) & 0x01FFFFFFu)  // global addr [56:32]
            | 0x80000000u;                        // type=2 ("image")
    i32x8 g1;
    g1[0] = (int)flags;                                           // mask/ds/pad
    g1[1] = (int)((tensor_d0 & 0xFFFFu) << 16);                   // bar=0 | td0.lo
    g1[2] = (int)((tensor_d0 >> 16) | ((tensor_d1 & 0xFFFFu) << 16));
    g1[3] = (int)((tensor_d1 >> 16) | (tile_d0 << 16));           // td1.hi | tile0
    g1[4] = (int)(tile_d1 & 0xFFFFu);                             // tile1, tile2=0
    g1[5] = (int)(unsigned)(stride0 & 0xFFFFFFFFu);               // stride0 lo
    g1[6] = (int)(unsigned)((stride0 >> 32) & 0xFFFFu);           // stride0 hi
    g1[7] = 0;
    i32x4 g2 = {0, 0, 0, 0};
    i32x4 g3 = {0, 0, 0, 0};
    asm volatile("tensor_load_to_lds %0, %1, %2, %3"
                 :: "s"(g0), "s"(g1), "s"(g2), "s"(g3)
                 : "memory");
}

// ---------------- elementwise helpers ----------------
__global__ void cvt_f32_to_bf16(const float* __restrict__ s, bf16* __restrict__ d, int n) {
    int i = blockIdx.x * blockDim.x + threadIdx.x;
    if (i < n) d[i] = f2bf(s[i]);
}

__global__ void zero_f32(float* __restrict__ p, int n) {
    int i = blockIdx.x * blockDim.x + threadIdx.x;
    if (i < n) p[i] = 0.0f;
}

// ---------------- generic 128x128x32 bf16 WMMA GEMM, TDM-staged ----------------
// 8 waves (wave32); wave (wr,wc) computes a 32x64 sub-tile = 2x4 WMMA tiles.
// A tile 128x32 staged via TDM with pad 1 dword / 16 dwords -> As[128][34]
// B tile  32x128 staged via TDM with pad 2 dwords / 64 dwords -> Bs[32][132]
// EPI: 0 = store bf16 (qkv)
//      1 = r = resid + 0.2*acc + bias ; store f32 + bf16   (w_o + residual1)
//      2 = relu(acc + bias) ; store bf16                   (ffn1)
//      3 = r = resid + acc + bias + bias2 ; store f32      (ffn2 + residual2)
template <int EPI>
__global__ __launch_bounds__(256) void gemm_bf16(
    const bf16* __restrict__ A, const bf16* __restrict__ B,
    bf16* __restrict__ Cb, float* __restrict__ Cf,
    const float* __restrict__ bias, const float* __restrict__ bias2,
    const float* __restrict__ resid,
    int M, int N, int K, int lda, int ldb)
{
    __shared__ __align__(16) bf16 As[128][34];
    __shared__ __align__(16) bf16 Bs[32][132];

    const int tid  = threadIdx.x;
    const int lane = tid & 31;
    const int wid  = tid >> 5;
    const int wr   = wid >> 1;      // 0..3
    const int wc   = wid & 1;       // 0..1
    const int rowbase = blockIdx.y * 128;
    const int colbase = blockIdx.x * 128;

    v8f acc[2][4];
    v8f zv = {};
    for (int mi = 0; mi < 2; ++mi)
        for (int ni = 0; ni < 4; ++ni) acc[mi][ni] = zv;

    const unsigned ldsA = lds_off_of(&As[0][0]);
    const unsigned ldsB = lds_off_of(&Bs[0][0]);

    for (int k0 = 0; k0 < K; k0 += 32) {
        if (wid == 0) {            // A tile: 32 wide x 128 rows, stride lda
            tdm_load_2d(ldsA, &A[(size_t)rowbase * lda + k0],
                        TDM_FLAGS(3u, 0u), 32, 128, 32, 128,
                        (unsigned long long)lda);
            __builtin_amdgcn_s_wait_tensorcnt(0);
        } else if (wid == 1) {     // B tile: 128 wide x 32 rows, stride ldb
            tdm_load_2d(ldsB, &B[(size_t)k0 * ldb + colbase],
                        TDM_FLAGS(5u, 1u), 128, 32, 128, 32,
                        (unsigned long long)ldb);
            __builtin_amdgcn_s_wait_tensorcnt(0);
        }
        __syncthreads();

        v16bf af[2], bfr[4];
        for (int mi = 0; mi < 2; ++mi) {
            int m = wr * 32 + mi * 16 + (lane & 15);
            for (int e = 0; e < 16; ++e) {
                int kk = ((e >> 3) << 4) + ((lane >> 4) << 3) + (e & 7);
                af[mi][e] = As[m][kk];
            }
        }
        for (int ni = 0; ni < 4; ++ni) {
            int n = wc * 64 + ni * 16 + (lane & 15);
            for (int e = 0; e < 16; ++e) {
                int kk = ((e >> 3) << 4) + ((lane >> 4) << 3) + (e & 7);
                bfr[ni][e] = Bs[kk][n];
            }
        }
        for (int mi = 0; mi < 2; ++mi)
            for (int ni = 0; ni < 4; ++ni)
                acc[mi][ni] = __builtin_amdgcn_wmma_f32_16x16x32_bf16(
                    false, af[mi], false, bfr[ni], (short)0, acc[mi][ni], false, false);
        __syncthreads();
    }

    // epilogue (C layout: VGPR j -> M = j + 8*(lane/16), N = lane%16)
    for (int mi = 0; mi < 2; ++mi)
        for (int ni = 0; ni < 4; ++ni)
            for (int j = 0; j < 8; ++j) {
                int m = rowbase + wr * 32 + mi * 16 + j + ((lane >> 4) << 3);
                int n = colbase + wc * 64 + ni * 16 + (lane & 15);
                float v = acc[mi][ni][j];
                size_t idx = (size_t)m * N + n;
                if constexpr (EPI == 0) {
                    Cb[idx] = f2bf(v);
                } else if constexpr (EPI == 1) {
                    float r = resid[idx] + 0.2f * v + bias[n];
                    Cf[idx] = r; Cb[idx] = f2bf(r);
                } else if constexpr (EPI == 2) {
                    v += bias[n]; v = v > 0.0f ? v : 0.0f;
                    Cb[idx] = f2bf(v);
                } else {
                    float r = resid[idx] + v + bias[n] + bias2[n];
                    Cf[idx] = r;
                }
            }
}

// ---------------- attention scores: w = (q.k^T/(8*tau))^2 * mask ----------------
// grid: (coltile 8, rowtile 8, bh 48). Q/K tiles staged by TDM (64 wide x 128
// rows, pad 1 dword / 32 dwords -> [128][66]). Writes bf16 scores + rowsums.
__global__ __launch_bounds__(256) void attn_scores(
    const bf16* __restrict__ qkvb, const float* __restrict__ mask,
    const float* __restrict__ tau,
    bf16* __restrict__ wsc, float* __restrict__ rowsum)
{
    __shared__ __align__(16) bf16 Qs[128][66];
    __shared__ __align__(16) bf16 Ks[128][66];
    __shared__ float rs[128];

    const int tid  = threadIdx.x;
    const int lane = tid & 31;
    const int wid  = tid >> 5;
    const int wr   = wid >> 1, wc = wid & 1;
    const int colbase = blockIdx.x * 128;
    const int rowbase = blockIdx.y * 128;
    const int bh = blockIdx.z;
    const int b  = bh / HEADS, h = bh % HEADS;
    const float sc = 0.125f / tau[h];    // 1/sqrt(64)/tau

    if (wid == 0) {
        tdm_load_2d(lds_off_of(&Qs[0][0]),
                    &qkvb[(size_t)(b * TSEQ + rowbase) * (3 * HDIM) + h * HD],
                    TDM_FLAGS(4u, 0u), 64, 128, 64, 128, 3 * HDIM);
        __builtin_amdgcn_s_wait_tensorcnt(0);
    } else if (wid == 1) {
        tdm_load_2d(lds_off_of(&Ks[0][0]),
                    &qkvb[(size_t)(b * TSEQ + colbase) * (3 * HDIM) + HDIM + h * HD],
                    TDM_FLAGS(4u, 0u), 64, 128, 64, 128, 3 * HDIM);
        __builtin_amdgcn_s_wait_tensorcnt(0);
    }
    if (tid < 128) rs[tid] = 0.0f;
    __syncthreads();

    v8f acc[2][4];
    v8f zv = {};
    for (int mi = 0; mi < 2; ++mi)
        for (int ni = 0; ni < 4; ++ni) acc[mi][ni] = zv;

    for (int ks = 0; ks < 2; ++ks) {          // K = 64 -> two 32-wide steps
        v16bf af[2], bfr[4];
        for (int mi = 0; mi < 2; ++mi) {
            int m = wr * 32 + mi * 16 + (lane & 15);
            for (int e = 0; e < 16; ++e) {
                int kk = ks * 32 + ((e >> 3) << 4) + ((lane >> 4) << 3) + (e & 7);
                af[mi][e] = Qs[m][kk];
            }
        }
        for (int ni = 0; ni < 4; ++ni) {
            int n = wc * 64 + ni * 16 + (lane & 15);
            for (int e = 0; e < 16; ++e) {
                int kk = ks * 32 + ((e >> 3) << 4) + ((lane >> 4) << 3) + (e & 7);  // B[k][n] = K[n][k]
                bfr[ni][e] = Ks[n][kk];
            }
        }
        for (int mi = 0; mi < 2; ++mi)
            for (int ni = 0; ni < 4; ++ni)
                acc[mi][ni] = __builtin_amdgcn_wmma_f32_16x16x32_bf16(
                    false, af[mi], false, bfr[ni], (short)0, acc[mi][ni], false, false);
    }

    for (int mi = 0; mi < 2; ++mi)
        for (int ni = 0; ni < 4; ++ni)
            for (int j = 0; j < 8; ++j) {
                int ml = wr * 32 + mi * 16 + j + ((lane >> 4) << 3);
                int nl = wc * 64 + ni * 16 + (lane & 15);
                float s = acc[mi][ni][j] * sc;
                float w = s * s * mask[(size_t)b * TSEQ + colbase + nl];
                atomicAdd(&rs[ml], w);
                wsc[((size_t)bh * TSEQ + rowbase + ml) * TSEQ + colbase + nl] = f2bf(w);
            }
    __syncthreads();
    if (tid < 128)
        atomicAdd(&rowsum[(size_t)bh * TSEQ + rowbase + tid], rs[tid]);
}

// ---------------- attention out: (w / (rowsum+eps)) @ v ----------------
// grid: (rowtile 8, bh 48). V tile via TDM; score tile staged manually so the
// 1/(sum+eps) scaling happens in-flight. Each of 8 waves owns a 16x64 slab.
__global__ __launch_bounds__(256) void attn_out(
    const bf16* __restrict__ qkvb, const bf16* __restrict__ wsc,
    const float* __restrict__ rowsum, bf16* __restrict__ attnb)
{
    __shared__ __align__(16) bf16 Ws[128][34];
    __shared__ __align__(16) bf16 Vs[32][66];
    __shared__ float invs[128];

    const int tid  = threadIdx.x;
    const int lane = tid & 31;
    const int wid  = tid >> 5;
    const int rowbase = blockIdx.x * 128;
    const int bh = blockIdx.y;
    const int b  = bh / HEADS, h = bh % HEADS;

    if (tid < 128)
        invs[tid] = 1.0f / (rowsum[(size_t)bh * TSEQ + rowbase + tid] + EPSV);
    __syncthreads();

    v8f acc[4];
    v8f zv = {};
    for (int ni = 0; ni < 4; ++ni) acc[ni] = zv;

    for (int k0 = 0; k0 < TSEQ; k0 += 32) {
        if (wid == 1) {            // V tile: 64 wide x 32 rows -> Vs[32][66]
            tdm_load_2d(lds_off_of(&Vs[0][0]),
                        &qkvb[(size_t)(b * TSEQ + k0) * (3 * HDIM) + 2 * HDIM + h * HD],
                        TDM_FLAGS(4u, 0u), 64, 32, 64, 32, 3 * HDIM);
            __builtin_amdgcn_s_wait_tensorcnt(0);
        }
        for (int c = tid; c < 512; c += 256) {      // Ws 128x32, scaled in-flight
            int row = c >> 2, seg = c & 3;
            uint4 d = *reinterpret_cast<const uint4*>(
                &wsc[((size_t)bh * TSEQ + rowbase + row) * TSEQ + k0 + seg * 8]);
            const bf16* e = reinterpret_cast<const bf16*>(&d);
            float s = invs[row];
            bf16* dst = &Ws[row][seg * 8];
            for (int i = 0; i < 8; ++i) dst[i] = f2bf(bf2f(e[i]) * s);
        }
        if (k0 + 32 < TSEQ)
            __builtin_prefetch(&wsc[((size_t)bh * TSEQ + rowbase + (tid >> 1)) * TSEQ + k0 + 32], 0, 0);
        __syncthreads();

        v16bf af, bfr[4];
        int rw = wid * 16;
        for (int e = 0; e < 16; ++e) {
            int kk = ((e >> 3) << 4) + ((lane >> 4) << 3) + (e & 7);
            af[e] = Ws[rw + (lane & 15)][kk];
        }
        for (int ni = 0; ni < 4; ++ni)
            for (int e = 0; e < 16; ++e) {
                int kk = ((e >> 3) << 4) + ((lane >> 4) << 3) + (e & 7);
                bfr[ni][e] = Vs[kk][ni * 16 + (lane & 15)];
            }
        for (int ni = 0; ni < 4; ++ni)
            acc[ni] = __builtin_amdgcn_wmma_f32_16x16x32_bf16(
                false, af, false, bfr[ni], (short)0, acc[ni], false, false);
        __syncthreads();
    }

    for (int ni = 0; ni < 4; ++ni)
        for (int j = 0; j < 8; ++j) {
            int r = wid * 16 + j + ((lane >> 4) << 3);
            int c = ni * 16 + (lane & 15);
            attnb[(size_t)(b * TSEQ + rowbase + r) * HDIM + h * HD + c] = f2bf(acc[ni][j]);
        }
}

// ---------------- launch ----------------
extern "C" void kernel_launch(void* const* d_in, const int* in_sizes, int n_in,
                              void* d_out, int out_size, void* d_ws, size_t ws_size,
                              hipStream_t stream)
{
    (void)in_sizes; (void)n_in; (void)out_size; (void)ws_size;
    const float* x     = (const float*)d_in[0];
    const float* mask  = (const float*)d_in[1];
    const float* w_qkv = (const float*)d_in[2];
    const float* w_o   = (const float*)d_in[3];
    const float* tau   = (const float*)d_in[4];
    const float* w1    = (const float*)d_in[5];
    const float* b1    = (const float*)d_in[6];
    const float* w2    = (const float*)d_in[7];
    const float* b2    = (const float*)d_in[8];
    const float* nb1   = (const float*)d_in[9];
    const float* nb2   = (const float*)d_in[10];
    float* out = (float*)d_out;

    char* base = (char*)d_ws;
    size_t cur = 0;
    auto alloc = [&](size_t bytes) -> char* {
        cur = (cur + 255) & ~(size_t)255;
        char* p = base + cur; cur += bytes; return p;
    };
    bf16*  xb     = (bf16*) alloc((size_t)MTOK * HDIM * 2);
    bf16*  wqkvb  = (bf16*) alloc((size_t)HDIM * 3 * HDIM * 2);
    bf16*  wob    = (bf16*) alloc((size_t)HDIM * HDIM * 2);
    bf16*  w1b    = (bf16*) alloc((size_t)HDIM * INTER * 2);
    bf16*  w2b    = (bf16*) alloc((size_t)INTER * HDIM * 2);
    bf16*  qkvb   = (bf16*) alloc((size_t)MTOK * 3 * HDIM * 2);
    bf16*  wsc    = (bf16*) alloc((size_t)BATCH * HEADS * TSEQ * TSEQ * 2);
    float* rsum   = (float*)alloc((size_t)BATCH * HEADS * TSEQ * 4);
    bf16*  attnb  = (bf16*) alloc((size_t)MTOK * HDIM * 2);
    float* x1f    = (float*)alloc((size_t)MTOK * HDIM * 4);
    bf16*  x1b    = (bf16*) alloc((size_t)MTOK * HDIM * 2);
    bf16*  h1b    = (bf16*) alloc((size_t)MTOK * INTER * 2);

    auto cvt = [&](const float* s, bf16* d, int n) {
        cvt_f32_to_bf16<<<(n + 255) / 256, 256, 0, stream>>>(s, d, n);
    };
    cvt(x,     xb,    MTOK * HDIM);
    cvt(w_qkv, wqkvb, HDIM * 3 * HDIM);
    cvt(w_o,   wob,   HDIM * HDIM);
    cvt(w1,    w1b,   HDIM * INTER);
    cvt(w2,    w2b,   INTER * HDIM);

    int nrs = BATCH * HEADS * TSEQ;
    zero_f32<<<(nrs + 255) / 256, 256, 0, stream>>>(rsum, nrs);

    // qkv = x @ w_qkv   -> bf16 [4096, 2304]
    gemm_bf16<0><<<dim3(3 * HDIM / 128, MTOK / 128), 256, 0, stream>>>(
        xb, wqkvb, qkvb, nullptr, nullptr, nullptr, nullptr,
        MTOK, 3 * HDIM, HDIM, HDIM, 3 * HDIM);

    // attention scores + rowsums
    attn_scores<<<dim3(TSEQ / 128, TSEQ / 128, BATCH * HEADS), 256, 0, stream>>>(
        qkvb, mask, tau, wsc, rsum);

    // attn = (w / rowsum) @ v  -> bf16 [4096, 768]
    attn_out<<<dim3(TSEQ / 128, BATCH * HEADS), 256, 0, stream>>>(
        qkvb, wsc, rsum, attnb);

    // x1 = x + 0.2*(attn @ w_o) + nb1  -> f32 + bf16
    gemm_bf16<1><<<dim3(HDIM / 128, MTOK / 128), 256, 0, stream>>>(
        attnb, wob, x1b, x1f, nb1, nullptr, x,
        MTOK, HDIM, HDIM, HDIM, HDIM);

    // h1 = relu(x1 @ w1 + b1) -> bf16 [4096, 3072]
    gemm_bf16<2><<<dim3(INTER / 128, MTOK / 128), 256, 0, stream>>>(
        x1b, w1b, h1b, nullptr, b1, nullptr, nullptr,
        MTOK, INTER, HDIM, HDIM, INTER);

    // out = x1 + (h1 @ w2) + b2 + nb2 -> f32
    gemm_bf16<3><<<dim3(HDIM / 128, MTOK / 128), 256, 0, stream>>>(
        h1b, w2b, nullptr, out, b2, nb2, x1f,
        MTOK, HDIM, INTER, INTER, HDIM);
}
